// GraphConvolutionalLayer_44246753084001
// MI455X (gfx1250) — matrix operand order repaired and checked by reading, not compiled, not run on hardware
//
#include <hip/hip_runtime.h>
#include <hip/hip_bf16.h>

// GCN layer: out = adj @ (input @ W^T) + b
//   N=16384, IN_F=128, OUT_F=64
//
// Roofline: adj = 1 GiB fp32 read once -> >=46us @ 23.3 TB/s.
// Intensity = 32 FLOP/B -> need ~750 TFLOP/s to stay memory bound;
// fp32 WMMA (16x16x4) is 1/8 the MACs/instr of bf16 (16x16x32), so we use
// bf16 WMMA with a 2-term hi/lo split (3 WMMAs) for ~fp32 accuracy at 3/8
// of the fp32-WMMA instruction cost -> memory bound.

#define NN   16384
#define INF  128
#define OUTF 64

typedef __bf16 v16bf __attribute__((ext_vector_type(16)));
typedef float  v8f   __attribute__((ext_vector_type(8)));
typedef float  v4f   __attribute__((ext_vector_type(4)));

// ---------------------------------------------------------------------------
// Kernel 1: h[m][n] = sum_k input[m][k] * W[n][k]   (fp32, tiny GEMM)
// block = 256 threads: 64 n-columns x 4 m-rows; W staged in LDS (32 KB).
// ---------------------------------------------------------------------------
__global__ void __launch_bounds__(256)
gcn_h_linear(const float* __restrict__ inp, const float* __restrict__ W,
             float* __restrict__ h) {
  __shared__ float Wl[OUTF * INF];              // 32 KB
  const int tid = threadIdx.x;
  const v4f* W4  = (const v4f*)W;               // 2048 x float4
  v4f*       Wl4 = (v4f*)Wl;
#pragma unroll
  for (int i = 0; i < 8; ++i) Wl4[tid + 256 * i] = W4[tid + 256 * i];
  __syncthreads();

  const int n = tid & 63;
  const int m = blockIdx.x * 4 + (tid >> 6);
  const float* __restrict__ row = inp + (size_t)m * INF;
  const float* __restrict__ wr  = Wl + n * INF;
  float acc = 0.f;
#pragma unroll 8
  for (int k = 0; k < INF; ++k) acc = fmaf(row[k], wr[k], acc);
  h[(size_t)m * OUTF + n] = acc;
}

// ---------------------------------------------------------------------------
// Kernel 2: split h into bf16 hi/lo and pre-pack into the WMMA B-fragment
// lane layout (ISA 7.12.2, 16-bit B 32x16):
//   lane l<16 : element e -> (K = e<8 ? e : 8+e,     N = l)
//   lane l>=16: element e -> (K = e<8 ? 8+e : 16+e,  N = l-16)
// One thread per (K-chunk c, N-tile t, lane l); flat id = ((c*4+t)*32+l),
// each thread emits 16 contiguous bf16 (32 B) -> main kernel loads b128s.
// ---------------------------------------------------------------------------
__global__ void __launch_bounds__(256)
gcn_pack_h(const float* __restrict__ h, __bf16* __restrict__ hi,
           __bf16* __restrict__ lo) {
  const int gid  = blockIdx.x * 256 + threadIdx.x;   // 65536 total
  const int l    = gid & 31;
  const int t    = (gid >> 5) & 3;
  const int c    = gid >> 7;                          // 0..511
  const int n    = t * 16 + (l & 15);
  const int kb   = c * 32 + ((l < 16) ? 0 : 8);
  const size_t base = (size_t)gid * 16;
#pragma unroll
  for (int e = 0; e < 16; ++e) {
    float x   = h[(size_t)(kb + e + ((e < 8) ? 0 : 8)) * OUTF + n];
    __bf16 xh = (__bf16)x;                            // RNE hi part
    float  r  = x - (float)xh;                        // exact residual
    hi[base + e] = xh;
    lo[base + e] = (__bf16)r;                         // |err| ~ 2^-18 |x|
  }
}

// ---------------------------------------------------------------------------
// Kernel 3: out = adj @ h + b via v_wmma_f32_16x16x32_bf16.
// Block = 256 thr = 8 waves: wave w -> (m-tile = w&1, K-quarter = w>>1).
// Each wave: 16 rows x K=4096, 128 chunks of K=32; per chunk:
//   A: 4 nontemporal b128 fp32 loads/lane -> hi/lo split in regs
//   B: 8 prepacked v16bf fragments (hi/lo x 4 N-tiles) from L2/L1
//   12 WMMAs (4 N-tiles x {ahi*bhi, alo*bhi, ahi*blo})
// K-quarters combined deterministically via LDS; bias added at the end.
// ---------------------------------------------------------------------------
__global__ void __launch_bounds__(256)
gcn_aggregate(const float* __restrict__ adj, const __bf16* __restrict__ Bhi,
              const __bf16* __restrict__ Blo, const float* __restrict__ bias,
              float* __restrict__ out) {
  __shared__ float cbuf[2][4][16][OUTF];              // 32 KB
  const int tid  = threadIdx.x;
  const int lane = tid & 31;
  const int w    = tid >> 5;                          // 0..7
  const int mt   = w & 1;                             // m-tile in block
  const int kq   = w >> 1;                            // 0..3 K-quarter
  const int m0   = blockIdx.x * 32 + mt * 16;
  const int row  = m0 + (lane & 15);
  const int kh   = (lane < 16) ? 0 : 8;               // A-fragment lane shift

  // A layout (ISA 7.12.2, 16-bit A 16x32): lane holds K=[kh,kh+8) and [kh+16,kh+24)
  const float* __restrict__ aptr =
      adj + (size_t)row * NN + (size_t)(kq * 4096 + kh);
  const __bf16* __restrict__ bh0 = Bhi + (size_t)(kq * 128) * 2048 + lane * 16;
  const __bf16* __restrict__ bl0 = Blo + (size_t)(kq * 128) * 2048 + lane * 16;

  v8f acc0 = {}, acc1 = {}, acc2 = {}, acc3 = {};

#pragma unroll 2
  for (int it = 0; it < 128; ++it) {
    const v4f* a4 = (const v4f*)(aptr + it * 32);
    v4f q0 = __builtin_nontemporal_load(a4 + 0);      // K kh+0..3
    v4f q1 = __builtin_nontemporal_load(a4 + 1);      // K kh+4..7
    v4f q2 = __builtin_nontemporal_load(a4 + 4);      // K kh+16..19
    v4f q3 = __builtin_nontemporal_load(a4 + 5);      // K kh+20..23

    float aq[16] = {q0[0], q0[1], q0[2], q0[3], q1[0], q1[1], q1[2], q1[3],
                    q2[0], q2[1], q2[2], q2[3], q3[0], q3[1], q3[2], q3[3]};
    v16bf ahi, alo;
#pragma unroll
    for (int e = 0; e < 16; ++e) {
      __bf16 hh = (__bf16)aq[e];
      ahi[e] = hh;
      alo[e] = (__bf16)(aq[e] - (float)hh);
    }

    const __bf16* bh = bh0 + (size_t)it * 2048;       // chunk stride: 4*32*16
    const __bf16* bl = bl0 + (size_t)it * 2048;
    v16bf b0h = *(const v16bf*)(bh);                  // N-tile stride: 512
    v16bf b1h = *(const v16bf*)(bh + 512);
    v16bf b2h = *(const v16bf*)(bh + 1024);
    v16bf b3h = *(const v16bf*)(bh + 1536);
    v16bf b0l = *(const v16bf*)(bl);
    v16bf b1l = *(const v16bf*)(bl + 512);
    v16bf b2l = *(const v16bf*)(bl + 1024);
    v16bf b3l = *(const v16bf*)(bl + 1536);

    acc0 = __builtin_amdgcn_wmma_f32_16x16x32_bf16(false, ahi, false, b0h, (short)0, acc0, false, false);
    acc1 = __builtin_amdgcn_wmma_f32_16x16x32_bf16(false, ahi, false, b1h, (short)0, acc1, false, false);
    acc2 = __builtin_amdgcn_wmma_f32_16x16x32_bf16(false, ahi, false, b2h, (short)0, acc2, false, false);
    acc3 = __builtin_amdgcn_wmma_f32_16x16x32_bf16(false, ahi, false, b3h, (short)0, acc3, false, false);
    acc0 = __builtin_amdgcn_wmma_f32_16x16x32_bf16(false, alo, false, b0h, (short)0, acc0, false, false);
    acc1 = __builtin_amdgcn_wmma_f32_16x16x32_bf16(false, alo, false, b1h, (short)0, acc1, false, false);
    acc2 = __builtin_amdgcn_wmma_f32_16x16x32_bf16(false, alo, false, b2h, (short)0, acc2, false, false);
    acc3 = __builtin_amdgcn_wmma_f32_16x16x32_bf16(false, alo, false, b3h, (short)0, acc3, false, false);
    acc0 = __builtin_amdgcn_wmma_f32_16x16x32_bf16(false, ahi, false, b0l, (short)0, acc0, false, false);
    acc1 = __builtin_amdgcn_wmma_f32_16x16x32_bf16(false, ahi, false, b1l, (short)0, acc1, false, false);
    acc2 = __builtin_amdgcn_wmma_f32_16x16x32_bf16(false, ahi, false, b2l, (short)0, acc2, false, false);
    acc3 = __builtin_amdgcn_wmma_f32_16x16x32_bf16(false, ahi, false, b3l, (short)0, acc3, false, false);
  }

  // C/D layout: lane<16 -> (M=v, N=lane); lane>=16 -> (M=8+v, N=lane-16)
  const int nlo  = lane & 15;
  const int mrow = (lane < 16) ? 0 : 8;
#pragma unroll
  for (int v = 0; v < 8; ++v) {
    cbuf[mt][kq][mrow + v][0  + nlo] = acc0[v];
    cbuf[mt][kq][mrow + v][16 + nlo] = acc1[v];
    cbuf[mt][kq][mrow + v][32 + nlo] = acc2[v];
    cbuf[mt][kq][mrow + v][48 + nlo] = acc3[v];
  }
  __syncthreads();

  // deterministic K-quarter reduction + bias: 2048 outputs / 256 threads
#pragma unroll
  for (int i = 0; i < 8; ++i) {
    const int idx   = tid + 256 * i;
    const int n     = idx & 63;
    const int mm    = (idx >> 6) & 15;
    const int mtile = idx >> 10;
    float s = cbuf[mtile][0][mm][n] + cbuf[mtile][1][mm][n] +
              cbuf[mtile][2][mm][n] + cbuf[mtile][3][mm][n];
    out[(size_t)(blockIdx.x * 32 + mtile * 16 + mm) * OUTF + n] = s + bias[n];
  }
}

// ---------------------------------------------------------------------------
extern "C" void kernel_launch(void* const* d_in, const int* in_sizes, int n_in,
                              void* d_out, int out_size, void* d_ws, size_t ws_size,
                              hipStream_t stream) {
  const float* inp  = (const float*)d_in[0];   // [16384,128]
  const float* adj  = (const float*)d_in[1];   // [16384,16384]
  const float* W    = (const float*)d_in[2];   // [64,128]
  const float* bias = (const float*)d_in[3];   // [64]
  float* out = (float*)d_out;                  // [16384,64]

  // Workspace layout (8 MiB total):
  //   [0, 4MiB)    h fp32
  //   [4, 6MiB)    packed h hi (bf16)
  //   [6, 8MiB)    packed h lo (bf16)
  char* ws = (char*)d_ws;
  float*  h   = (float*)ws;
  __bf16* Bhi = (__bf16*)(ws + (size_t)4 * 1024 * 1024);
  __bf16* Blo = (__bf16*)(ws + (size_t)6 * 1024 * 1024);

  gcn_h_linear<<<NN / 4, 256, 0, stream>>>(inp, W, h);
  gcn_pack_h<<<(512 * 4 * 32) / 256, 256, 0, stream>>>(h, Bhi, Blo);
  gcn_aggregate<<<NN / 32, 256, 0, stream>>>(adj, Bhi, Blo, bias, out);
}
